// KDALayer_39144331935947
// MI455X (gfx1250) — compile-verified, matrix operand order checked by reading, not run on hardware
//
#include <hip/hip_runtime.h>
#include <hip/hip_bf16.h>

// ---------------- problem constants ----------------
constexpr int Bc     = 2;
constexpr int Sc     = 2048;
constexpr int DMc    = 2048;
constexpr int Hc     = 8;
constexpr int DHc    = 128;
constexpr int INNERc = Hc * DHc;   // 1024
constexpr int CKc    = 64;         // attention chunk
constexpr int BSc    = Bc * Sc;    // 4096 rows

typedef __attribute__((ext_vector_type(16))) __bf16        v16bf;
typedef __attribute__((ext_vector_type(8)))  float         v8f;
typedef __attribute__((ext_vector_type(4)))  unsigned int  v4u;
typedef __attribute__((ext_vector_type(8)))  int           v8i;
typedef __attribute__((ext_vector_type(4)))  int           v4i;

static __device__ __forceinline__ v8f wmma_bf16(v16bf a, v16bf b, v8f c) {
  // D = A x B + C, 16x16x32 bf16 -> f32
  return __builtin_amdgcn_wmma_f32_16x16x32_bf16(false, a, false, b, (short)0, c,
                                                 false, false);
}

// Load a 16x32 bf16 fragment: lane L (m = L&15, half = L>>4) reads 16
// contiguous bf16 along K from row (tile*16 + m), K-offset kk*32 + half*16.
static __device__ __forceinline__ v16bf ldfrag(const __bf16* base, int stride,
                                               int tile, int kk, int lane) {
  int hf = lane >> 4;
  int m  = lane & 15;
  return *(const v16bf*)(base + (size_t)(tile * 16 + m) * stride + kk * 32 + hf * 16);
}

static __device__ __forceinline__ v8f vzero8() {
  v8f z = {0.f, 0.f, 0.f, 0.f, 0.f, 0.f, 0.f, 0.f};
  return z;
}

// ---- Tensor Data Mover: 2D bf16 tile (tile_d0 elems x tile_d1 rows) from
// global (row pitch stride_elems, in 2-byte elements) into LDS at lds_off.
// D# per CDNA5 ISA 8.3/8.4: group0 = {count/type/addr/lds}, group1 = dims.
// Toolchain here uses the 6-arg builtin: (v4u, v8i, v4i, v4i, v8i, cpol).
static __device__ __forceinline__ void tdm_load_2d_bf16(
    unsigned lds_off, const void* gptr, unsigned tensor_d0, unsigned tensor_d1,
    unsigned tile_d0, unsigned tile_d1, unsigned stride_elems) {
  unsigned long long ga = (unsigned long long)gptr;
  v4u g0 = {1u,                                    // count=1 (valid), user mode
            lds_off,                               // lds_addr (bytes)
            (unsigned)ga,                          // global_addr[31:0]
            (unsigned)((ga >> 32) & 0x01FFFFFFu)   // global_addr[56:32]
                | 0x80000000u};                    // type=2 ("image")
  v8i g1 = {(int)(1u << 16),                       // data_size=1 -> 2 bytes
            (int)(tensor_d0 << 16),                // tensor_dim0[15:0] @bit48
            (int)((tensor_d0 >> 16) | (tensor_d1 << 16)),  // dim0 hi | dim1 lo
            (int)((tensor_d1 >> 16) | (tile_d0 << 16)),    // dim1 hi | tile_dim0
            (int)(tile_d1 & 0xFFFFu),              // tile_dim1 (tile_dim2=0)
            (int)stride_elems,                     // tensor_dim0_stride[31:0]
            0, 0};                                 // stride hi, dim1_stride=0
  v4i z4 = {0, 0, 0, 0};                           // 2D: groups 2/3 unused
  v8i z8 = {0, 0, 0, 0, 0, 0, 0, 0};
  __builtin_amdgcn_tensor_load_to_lds(g0, g1, z4, z4, z8, 0);
}

// ---------------- f32 -> bf16 convert ----------------
__global__ void __launch_bounds__(256)
cvt_f32_bf16(const float* __restrict__ in, __bf16* __restrict__ out, size_t n) {
  size_t i  = (size_t)blockIdx.x * 256 + threadIdx.x;
  size_t st = (size_t)gridDim.x * 256;
  for (; i < n; i += st) out[i] = (__bf16)in[i];
}

// ---------------- tiled WMMA GEMM: C[M,N] = A[M,K] * Bw[N,K]^T ----------------
// TDM-staged, double-buffered. A row-major (K contig), Bw row-major (K contig).
// Dynamic LDS layout (extern shared starts at LDS offset 0):
//   A buf0 @0, A buf1 @8192, B buf0 @16384, B buf1 @24576  (bytes)
constexpr size_t GEMM_SMEM = 4 * 128 * 32 * 2;   // 32 KB
__global__ void __launch_bounds__(256)
gemm_bf16_nt(const __bf16* __restrict__ A, const __bf16* __restrict__ Bw,
             float* __restrict__ C, int M, int N, int K) {
  extern __shared__ char gsm[];
  __bf16* As = (__bf16*)gsm;          // 2 x 128x32
  __bf16* Bs = As + 2 * 128 * 32;     // 2 x 128x32

  const int tid  = threadIdx.x;
  const int lane = tid & 31;
  const int w    = tid >> 5;
  const int hf   = lane >> 4;
  const int m15  = lane & 15;
  const int bn   = blockIdx.x * 128;
  const int bm   = blockIdx.y * 128;

  v8f acc[8];
  for (int c = 0; c < 8; ++c) acc[c] = vzero8();

  // prologue: DMA first K-slab into buffer 0 (wave 0 only; uniform branch)
  if (w == 0) {
    tdm_load_2d_bf16(0,     A  + (size_t)bm * K, 32, 4096, 32, 128, (unsigned)K);
    tdm_load_2d_bf16(16384, Bw + (size_t)bn * K, 32, 4096, 32, 128, (unsigned)K);
  }

  int buf = 0;
  for (int k0 = 0; k0 < K; k0 += 32, buf ^= 1) {
    if (w == 0) {
      if (k0 + 32 < K) {   // DMA next slab into other buffer, then wait cur<=done
        int nb = buf ^ 1;
        tdm_load_2d_bf16((unsigned)(nb * 8192),
                         A + (size_t)bm * K + k0 + 32, 32, 4096, 32, 128, (unsigned)K);
        tdm_load_2d_bf16((unsigned)(16384 + nb * 8192),
                         Bw + (size_t)bn * K + k0 + 32, 32, 4096, 32, 128, (unsigned)K);
        __builtin_amdgcn_s_wait_tensorcnt((short)2);
      } else {
        __builtin_amdgcn_s_wait_tensorcnt((short)0);
      }
    }
    __syncthreads();   // publish current buffer to all 8 waves

    const __bf16* Ab = As + buf * 4096;
    const __bf16* Bb = Bs + buf * 4096;
    v16bf a = ldfrag(Ab, 32, w, 0, lane);
    for (int c = 0; c < 8; ++c) {
      v16bf b = ldfrag(Bb, 32, c, 0, lane);
      acc[c] = wmma_bf16(a, b, acc[c]);
    }
    __syncthreads();   // all reads of `buf` done before it is DMA-overwritten
  }

  for (int c = 0; c < 8; ++c)
    for (int r = 0; r < 8; ++r)
      C[(size_t)(bm + w * 16 + r + hf * 8) * N + bn + c * 16 + m15] = acc[c][r];
}

// ---------------- alpha = sigmoid(alpha_log); la[h] = log(clip(mean)) --------
__global__ void __launch_bounds__(256)
alpha_kernel(const float* __restrict__ alpha_log, float* __restrict__ alpha,
             float* __restrict__ la) {
  const int t = threadIdx.x;
  for (int i = t; i < Hc * DHc; i += 256)
    alpha[i] = 1.f / (1.f + __expf(-alpha_log[i]));
  __syncthreads();
  const int w = t >> 5, lane = t & 31;   // wave w handles head w
  float s = 0.f;
  for (int d = lane; d < DHc; d += 32) s += alpha[w * DHc + d];
  for (int o = 16; o > 0; o >>= 1) s += __shfl_down(s, o, 32);
  if (lane == 0) la[w] = __logf(fmaxf(s * (1.f / DHc), 1e-6f));
}

// ---------------- beta[b,s,h] = sigmoid(x . Wb[h] + bb[h]) -------------------
__global__ void __launch_bounds__(256)
beta_kernel(const float* __restrict__ x, const float* __restrict__ Wb,
            const float* __restrict__ bb, float* __restrict__ beta) {
  const int ms   = blockIdx.x;          // b*S + s
  const int w    = threadIdx.x >> 5;    // head
  const int lane = threadIdx.x & 31;
  const float* xr = x  + (size_t)ms * DMc;
  const float* wr = Wb + (size_t)w  * DMc;
  float s = 0.f;
  for (int d = lane; d < DMc; d += 32) s += xr[d] * wr[d];
  for (int o = 16; o > 0; o >>= 1) s += __shfl_down(s, o, 32);
  if (lane == 0)
    beta[(size_t)ms * Hc + w] = 1.f / (1.f + __expf(-(s + bb[w])));
}

// ---------------- RoPE(q,k) + convert q,k,v to bf16 --------------------------
__global__ void __launch_bounds__(256)
rope_cvt_kernel(const float* __restrict__ qf, const float* __restrict__ kf,
                const float* __restrict__ vf, __bf16* __restrict__ qb,
                __bf16* __restrict__ kb, __bf16* __restrict__ vb) {
  size_t u = (size_t)blockIdx.x * 256 + threadIdx.x;
  if (u >= (size_t)Bc * Sc * Hc * (DHc / 2)) return;
  const int i    = (int)(u & 63);
  const int h    = (int)((u >> 6) & 7);
  const int sidx = (int)((u >> 9) & 2047);
  const int b    = (int)(u >> 20);
  const size_t base = (((size_t)b * Sc + sidx) * Hc + h) * DHc;

  const float invf = __powf(10000.f, -((float)(2 * i)) * (1.f / (float)DHc));
  float sv, cv;
  __sincosf((float)sidx * invf, &sv, &cv);

  const float q1 = qf[base + i], q2 = qf[base + 64 + i];
  qb[base + i]      = (__bf16)(q1 * cv - q2 * sv);
  qb[base + 64 + i] = (__bf16)(q2 * cv + q1 * sv);
  const float k1 = kf[base + i], k2 = kf[base + 64 + i];
  kb[base + i]      = (__bf16)(k1 * cv - k2 * sv);
  kb[base + 64 + i] = (__bf16)(k2 * cv + k1 * sv);
  vb[base + i]      = (__bf16)vf[base + i];
  vb[base + 64 + i] = (__bf16)vf[base + 64 + i];
}

// ---------------- fused chunked decay-attention + state recurrence -----------
// One workgroup per (b,h). Carries P[d,e] (scalar-decay KV state for the
// attention output) and Ast[d,e] (per-(h,d) alpha state, the returned state)
// in LDS f32; all GEMMs via WMMA bf16; q/k/v chunks staged by the TDM.
// LDS byte offsets (extern shared starts at 0): qs=0, ks=16384, vs=32768.
constexpr size_t ATTN_SMEM =
    (size_t)(64 * 128 * 3      // qs, ks, vs
             + 128 * 64 * 3    // vts, kdts, kats
             + 64 * 128        // qds
             + 64 * 64         // wbf
             + 128 * 128) * 2  // ptb (bf16)
    + (size_t)(128 * 128 * 2 + 64) * 4;   // Pst, Ast, betas (f32)

__global__ void __launch_bounds__(256)
attn_state_kernel(const __bf16* __restrict__ qb, const __bf16* __restrict__ kb,
                  const __bf16* __restrict__ vb, const float* __restrict__ beta,
                  const float* __restrict__ la_arr,
                  const float* __restrict__ alpha, __bf16* __restrict__ ob,
                  float* __restrict__ state_out) {
  extern __shared__ char smem[];
  __bf16* qs    = (__bf16*)smem;            // [64][128]   @ LDS byte 0
  __bf16* ks    = qs   + 64 * 128;          // [64][128]   @ 16384
  __bf16* vs    = ks   + 64 * 128;          // [64][128]   @ 32768
  __bf16* vts   = vs   + 64 * 128;          // [128][64]  v^T
  __bf16* kdts  = vts  + 128 * 64;          // [128][64]  (decay*beta*k)^T
  __bf16* kats  = kdts + 128 * 64;          // [128][64]  (alpha^..*beta*k)^T
  __bf16* qds   = kats + 128 * 64;          // [64][128]  decayed q
  __bf16* wbf   = qds  + 64 * 128;          // [64][64]   masked scores
  __bf16* ptb   = wbf  + 64 * 64;           // [128][128] P^T in bf16
  float*  Pst   = (float*)(ptb + 128 * 128);// [128][128]
  float*  Ast   = Pst + 128 * 128;          // [128][128]
  float*  betas = Ast + 128 * 128;          // [64]

  const int tid  = threadIdx.x;
  const int lane = tid & 31;
  const int w    = tid >> 5;
  const int hf   = lane >> 4;
  const int m15  = lane & 15;
  const int bh   = blockIdx.x;
  const int b    = bh / Hc, h = bh % Hc;

  const float la   = la_arr[h];
  const float dfac = __expf(la * (float)CKc);   // per-chunk scalar decay

  for (int i = tid; i < 128 * 128; i += 256) { Pst[i] = 0.f; Ast[i] = 0.f; }

  const size_t rowstride = (size_t)Hc * DHc;    // 1024
  const __bf16* qbase = qb + ((size_t)b * Sc) * rowstride + (size_t)h * DHc;
  const __bf16* kbase = kb + ((size_t)b * Sc) * rowstride + (size_t)h * DHc;
  const __bf16* vbase = vb + ((size_t)b * Sc) * rowstride + (size_t)h * DHc;
  __syncthreads();

  for (int c0 = 0; c0 < Sc; c0 += CKc) {
    // ---- TDM: stage 64x128 bf16 q/k/v chunks (row pitch 1024) into LDS ----
    if (w == 0) {
      tdm_load_2d_bf16(0u,     qbase + (size_t)c0 * rowstride, 128, 4096, 128, 64, 1024);
      tdm_load_2d_bf16(16384u, kbase + (size_t)c0 * rowstride, 128, 4096, 128, 64, 1024);
      tdm_load_2d_bf16(32768u, vbase + (size_t)c0 * rowstride, 128, 4096, 128, 64, 1024);
      __builtin_amdgcn_s_wait_tensorcnt((short)0);
    }
    if (tid < CKc) betas[tid] = beta[(size_t)(b * Sc + c0 + tid) * Hc + h];
    __syncthreads();

    // ---- build derived LDS operands ----
    for (int lin = tid; lin < 128 * 64; lin += 256) {   // d-major transposes
      int d = lin >> 6, j = lin & 63;
      float kv = (float)ks[j * 128 + d];
      float bj = betas[j];
      vts[lin]  = vs[j * 128 + d];
      kdts[lin] = (__bf16)(kv * __expf(la * (float)(63 - j)) * bj);
      kats[lin] = (__bf16)(kv * __powf(alpha[h * DHc + d], (float)(63 - j)) * bj);
    }
    for (int lin = tid; lin < 64 * 128; lin += 256) {   // q * e^{la(i+1)}
      int i = lin >> 7;
      qds[lin] = (__bf16)((float)qs[lin] * __expf(la * (float)(i + 1)));
    }
    for (int lin = tid; lin < 128 * 128; lin += 256) {  // P^T -> bf16
      int e = lin >> 7, d = lin & 127;
      ptb[lin] = (__bf16)Pst[d * 128 + e];
    }
    __syncthreads();

    // ---- GEMM1: scores = q @ k^T (64x64, K=128), + mask/decay/beta ----
    {
      const int tr  = w >> 1;
      const int tc0 = (w & 1) * 2;
      v8f a0 = vzero8(), a1 = vzero8();
      for (int kk = 0; kk < 4; ++kk) {
        v16bf a  = ldfrag(qs, 128, tr, kk, lane);
        v16bf b0 = ldfrag(ks, 128, tc0 + 0, kk, lane);
        v16bf b1 = ldfrag(ks, 128, tc0 + 1, kk, lane);
        a0 = wmma_bf16(a, b0, a0);
        a1 = wmma_bf16(a, b1, a1);
      }
      for (int c = 0; c < 2; ++c) {
        v8f acc = c ? a1 : a0;
        int tc = tc0 + c;
        for (int r = 0; r < 8; ++r) {
          int i = tr * 16 + r + hf * 8;
          int j = tc * 16 + m15;
          float f = (i >= j) ? __expf(la * (float)(i - j)) * betas[j] : 0.f;
          wbf[i * 64 + j] = (__bf16)(acc[r] * f);
        }
      }
    }
    __syncthreads();

    // ---- GEMM2: out = scores@v + qds@P  (64x128) ----
    {
      const int tr  = w >> 1;
      const int tcb = (w & 1) * 4;
      v8f acc[4];
      for (int c = 0; c < 4; ++c) acc[c] = vzero8();
      for (int kk = 0; kk < 2; ++kk) {                  // intra, K=64
        v16bf a = ldfrag(wbf, 64, tr, kk, lane);
        for (int c = 0; c < 4; ++c) {
          v16bf bfr = ldfrag(vts, 64, tcb + c, kk, lane);
          acc[c] = wmma_bf16(a, bfr, acc[c]);
        }
      }
      for (int kk = 0; kk < 4; ++kk) {                  // inter, K=128
        v16bf a = ldfrag(qds, 128, tr, kk, lane);
        for (int c = 0; c < 4; ++c) {
          v16bf bfr = ldfrag(ptb, 128, tcb + c, kk, lane);
          acc[c] = wmma_bf16(a, bfr, acc[c]);
        }
      }
      for (int c = 0; c < 4; ++c) {
        int e = (tcb + c) * 16 + m15;
        for (int r = 0; r < 8; ++r) {
          int i = tr * 16 + r + hf * 8;
          ob[(size_t)(b * Sc + c0 + i) * INNERc + h * DHc + e] = (__bf16)acc[c][r];
        }
      }
    }

    // ---- GEMM3: P = e^{64 la} P + (decayed k)^T @ v ----
    {
      v8f acc[8];
      for (int c = 0; c < 8; ++c) acc[c] = vzero8();
      for (int kk = 0; kk < 2; ++kk) {
        v16bf a = ldfrag(kdts, 64, w, kk, lane);
        for (int c = 0; c < 8; ++c) {
          v16bf bfr = ldfrag(vts, 64, c, kk, lane);
          acc[c] = wmma_bf16(a, bfr, acc[c]);
        }
      }
      for (int c = 0; c < 8; ++c)
        for (int r = 0; r < 8; ++r) {
          int d = w * 16 + r + hf * 8;
          int e = c * 16 + m15;
          Pst[d * 128 + e] = dfac * Pst[d * 128 + e] + acc[c][r];
        }
    }

    // ---- GEMM4: Ast = alpha_d^64 * Ast + (alpha-decayed k)^T @ v ----
    {
      v8f acc[8];
      for (int c = 0; c < 8; ++c) acc[c] = vzero8();
      for (int kk = 0; kk < 2; ++kk) {
        v16bf a = ldfrag(kats, 64, w, kk, lane);
        for (int c = 0; c < 8; ++c) {
          v16bf bfr = ldfrag(vts, 64, c, kk, lane);
          acc[c] = wmma_bf16(a, bfr, acc[c]);
        }
      }
      for (int r = 0; r < 8; ++r) {
        int d = w * 16 + r + hf * 8;
        float afac = __powf(alpha[h * DHc + d], (float)CKc);
        for (int c = 0; c < 8; ++c) {
          int e = c * 16 + m15;
          Ast[d * 128 + e] = afac * Ast[d * 128 + e] + acc[c][r];
        }
      }
    }
    __syncthreads();   // closes all reads before next chunk's TDM overwrite
  }

  const size_t so = (size_t)bh * DHc * DHc;
  for (int i = tid; i < 128 * 128; i += 256) state_out[so + i] = Ast[i];
}

// ---------------- host-side orchestration ----------------
extern "C" void kernel_launch(void* const* d_in, const int* in_sizes, int n_in,
                              void* d_out, int out_size, void* d_ws, size_t ws_size,
                              hipStream_t stream) {
  (void)in_sizes; (void)n_in; (void)out_size; (void)ws_size;
  const float* x   = (const float*)d_in[0];
  const float* Wq  = (const float*)d_in[1];
  const float* Wk  = (const float*)d_in[2];
  const float* Wv  = (const float*)d_in[3];
  const float* Wo  = (const float*)d_in[4];
  const float* Wb  = (const float*)d_in[5];
  const float* bb  = (const float*)d_in[6];
  const float* alg = (const float*)d_in[7];

  float* out       = (float*)d_out;                       // (B,S,DM)
  float* state_out = out + (size_t)Bc * Sc * DMc;         // (B,H,DH,DH)

  // ---- workspace carving (256B aligned) ----
  char* p = (char*)d_ws;
  auto carve = [&](size_t bytes) {
    char* r = p;
    p += (bytes + 255) & ~(size_t)255;
    return r;
  };
  __bf16* xb   = (__bf16*)carve((size_t)BSc * DMc * 2);
  __bf16* wqb  = (__bf16*)carve((size_t)INNERc * DMc * 2);
  __bf16* wkb  = (__bf16*)carve((size_t)INNERc * DMc * 2);
  __bf16* wvb  = (__bf16*)carve((size_t)INNERc * DMc * 2);
  __bf16* wob  = (__bf16*)carve((size_t)DMc * INNERc * 2);
  float*  qf   = (float*)carve((size_t)BSc * INNERc * 4);
  float*  kf   = (float*)carve((size_t)BSc * INNERc * 4);
  float*  vf   = (float*)carve((size_t)BSc * INNERc * 4);
  __bf16* qb2  = (__bf16*)carve((size_t)BSc * INNERc * 2);
  __bf16* kb2  = (__bf16*)carve((size_t)BSc * INNERc * 2);
  __bf16* vb2  = (__bf16*)carve((size_t)BSc * INNERc * 2);
  __bf16* obuf = (__bf16*)carve((size_t)BSc * INNERc * 2);
  float*  betw = (float*)carve((size_t)BSc * Hc * 4);
  float*  alpw = (float*)carve((size_t)Hc * DHc * 4);
  float*  law  = (float*)carve((size_t)Hc * 4);

  // 1) bf16 conversions
  cvt_f32_bf16<<<4096, 256, 0, stream>>>(x,  xb,  (size_t)BSc * DMc);
  cvt_f32_bf16<<<2048, 256, 0, stream>>>(Wq, wqb, (size_t)INNERc * DMc);
  cvt_f32_bf16<<<2048, 256, 0, stream>>>(Wk, wkb, (size_t)INNERc * DMc);
  cvt_f32_bf16<<<2048, 256, 0, stream>>>(Wv, wvb, (size_t)INNERc * DMc);
  cvt_f32_bf16<<<2048, 256, 0, stream>>>(Wo, wob, (size_t)DMc * INNERc);

  // 2) alpha / log-alpha-mean, beta gate
  alpha_kernel<<<1, 256, 0, stream>>>(alg, alpw, law);
  beta_kernel<<<BSc, 256, 0, stream>>>(x, Wb, bb, betw);

  // 3) QKV projections (TDM-staged double-buffered WMMA GEMM, f32 results)
  dim3 gqkv(INNERc / 128, BSc / 128);
  gemm_bf16_nt<<<gqkv, 256, GEMM_SMEM, stream>>>(xb, wqb, qf, BSc, INNERc, DMc);
  gemm_bf16_nt<<<gqkv, 256, GEMM_SMEM, stream>>>(xb, wkb, kf, BSc, INNERc, DMc);
  gemm_bf16_nt<<<gqkv, 256, GEMM_SMEM, stream>>>(xb, wvb, vf, BSc, INNERc, DMc);

  // 4) RoPE + convert to bf16
  rope_cvt_kernel<<<(Bc * Sc * Hc * (DHc / 2)) / 256, 256, 0, stream>>>(
      qf, kf, vf, qb2, kb2, vb2);

  // 5) fused decay attention + state recurrence (1 WG per (b,h))
  attn_state_kernel<<<Bc * Hc, 256, ATTN_SMEM, stream>>>(
      qb2, kb2, vb2, betw, law, alpw, obuf, state_out);

  // 6) output projection -> d_out
  dim3 gout(DMc / 128, BSc / 128);
  gemm_bf16_nt<<<gout, 256, GEMM_SMEM, stream>>>(obuf, wob, out, BSc, DMc, INNERc);
}
